// Encoder_26834955665660
// MI455X (gfx1250) — matrix operand order repaired and checked by reading, not compiled, not run on hardware
//
#include <hip/hip_runtime.h>
#include <hip/hip_bf16.h>

// ---------------------------------------------------------------------------
// GRU encoder for MI455X (gfx1250, wave32, WMMA bf16).
//
//   Phase 0: one-pass f32 -> bf16 pack of W_ih, W_hh, emb, h0 into d_ws.
//   Phase 1: gi = gather(emb) @ W_ih^T + b_ih: 16x64 tile per wave,
//            4 wmma chains, depth-2 pipelined loads (sched_group_barrier).
//   Phase 2: 512 serial fused steps: each wave = 32 rows x 16 cols x 3 gates
//            = 6 independent wmma chains, pipelined; h_t ping-pongs in bf16,
//            f32 h_t goes straight to d_out[t]. W_hh bf16 (6MB) L2-resident.
// ---------------------------------------------------------------------------

typedef __attribute__((ext_vector_type(16))) __bf16 bf16x16;
typedef __attribute__((ext_vector_type(8)))  __bf16 bf16x8;
typedef __attribute__((ext_vector_type(8)))  float  v8f;

#define HID   1024
#define EMB_D 1024
#define G3    3072                 // 3*HID
#define BATCH 64
#define SEQ   512
#define VOCAB 32000

#define WMMA_BF16(a, b, c) \
    __builtin_amdgcn_wmma_f32_16x16x32_bf16(false, (a), false, (b), (short)0, (c), false, false)

// Scheduling-group hints: keep each iteration's VMEM-read group ahead of the
// previous iteration's WMMA group -> double-buffered registers, partial waits.
#if defined(__has_builtin)
#if __has_builtin(__builtin_amdgcn_sched_group_barrier)
#define SCHED_VMEM_READS(n) __builtin_amdgcn_sched_group_barrier(0x020, (n), 0)
#define SCHED_WMMAS(n)      __builtin_amdgcn_sched_group_barrier(0x008, (n), 0)
#endif
#endif
#ifndef SCHED_VMEM_READS
#define SCHED_VMEM_READS(n)
#define SCHED_WMMAS(n)
#endif

// ---------------------------------------------------------------------------
// f32 -> bf16 truncating pack, 2 elements per thread (one v_perm_b32 each).
// ---------------------------------------------------------------------------
__global__ void __launch_bounds__(256)
cvt_bf16_kernel(const float2* __restrict__ src, unsigned* __restrict__ dst, long n2) {
    long i      = (long)blockIdx.x * blockDim.x + threadIdx.x;
    long stride = (long)gridDim.x * blockDim.x;
    for (; i < n2; i += stride) {
        float2   f  = src[i];
        unsigned u0 = __builtin_bit_cast(unsigned, f.x);
        unsigned u1 = __builtin_bit_cast(unsigned, f.y);
        dst[i] = (u1 & 0xffff0000u) | (u0 >> 16);
    }
}

// ---------------------------------------------------------------------------
// Fragment loaders: pure bf16 vector loads.
// A 16x32: lane&15 = M row; lane>=16 = K+16 half (16 contiguous bf16).
// ---------------------------------------------------------------------------
__device__ __forceinline__ bf16x16 load_a_bf(const __bf16* __restrict__ arow,
                                             int k0, int lane) {
    int hi = lane >> 4;
    return *(const bf16x16*)(arow + k0 + hi * 16);
}

__device__ __forceinline__ bf16x16 load_a_bf_rows(const __bf16* __restrict__ A,
                                                  int lda, int k0, int lane) {
    return load_a_bf(A + (long)(lane & 15) * lda, k0, lane);
}

// B 32x16 (KxN): lane&15 = N column, lane>=16 = K+16 half; W stored (N,K)
// row-major so each lane reads 16 contiguous bf16 (32 bytes).
__device__ __forceinline__ bf16x16 load_b_bf(const __bf16* __restrict__ W,
                                             int ldw, int n0, int k0, int lane) {
    int n  = lane & 15;
    int hi = lane >> 4;
    return *(const bf16x16*)(W + (long)(n0 + n) * ldw + k0 + hi * 16);
}

__device__ __forceinline__ float sigmoidf_(float x) {
    return 1.0f / (1.0f + __expf(-x));
}

// ---------------------------------------------------------------------------
// Phase 1: gi = gather(emb) @ W_ih^T + b_ih for rows = T*64 rows.
// Wave tile 16(M) x 64(N), 4 chains, pipelined depth 2. 8 waves/block.
// ---------------------------------------------------------------------------
__global__ void __launch_bounds__(256)
gi_kernel(const int* __restrict__ input,        // rows tokens (chunk base)
          const __bf16* __restrict__ embb,      // (VOCAB, 1024) bf16
          const __bf16* __restrict__ Wb,        // (3072, 1024) bf16
          const float* __restrict__ b_ih,       // (3072,)
          float* __restrict__ gi)               // (rows, 3072) f32
{
    const int TN = G3 / 64;                     // 48 wave-tiles along N
    int  wave = threadIdx.x >> 5;
    int  lane = threadIdx.x & 31;
    long tile = (long)blockIdx.x * 8 + wave;
    int  tm   = (int)(tile / TN);
    int  tn   = (int)(tile % TN);
    int  row0 = tm * 16;
    int  n0   = tn * 64;

    int token = input[row0 + (lane & 15)];
    const __bf16* arow = embb + (long)token * EMB_D;

    v8f c0 = {}, c1 = {}, c2 = {}, c3 = {};

    // pipeline prologue
    bf16x16 a  = load_a_bf(arow, 0, lane);
    bf16x16 b0 = load_b_bf(Wb, EMB_D, n0,      0, lane);
    bf16x16 b1 = load_b_bf(Wb, EMB_D, n0 + 16, 0, lane);
    bf16x16 b2 = load_b_bf(Wb, EMB_D, n0 + 32, 0, lane);
    bf16x16 b3 = load_b_bf(Wb, EMB_D, n0 + 48, 0, lane);

#pragma unroll 2
    for (int k0 = 0; k0 < EMB_D - 32; k0 += 32) {
        int kn = k0 + 32;
        bf16x16 na  = load_a_bf(arow, kn, lane);
        bf16x16 nb0 = load_b_bf(Wb, EMB_D, n0,      kn, lane);
        bf16x16 nb1 = load_b_bf(Wb, EMB_D, n0 + 16, kn, lane);
        bf16x16 nb2 = load_b_bf(Wb, EMB_D, n0 + 32, kn, lane);
        bf16x16 nb3 = load_b_bf(Wb, EMB_D, n0 + 48, kn, lane);
        c0 = WMMA_BF16(a, b0, c0);
        c1 = WMMA_BF16(a, b1, c1);
        c2 = WMMA_BF16(a, b2, c2);
        c3 = WMMA_BF16(a, b3, c3);
        a = na; b0 = nb0; b1 = nb1; b2 = nb2; b3 = nb3;
        SCHED_VMEM_READS(10);   // next iter's 10 b128 loads first ...
        SCHED_WMMAS(4);         // ... then this iter's 4 wmmas
    }
    c0 = WMMA_BF16(a, b0, c0);
    c1 = WMMA_BF16(a, b1, c1);
    c2 = WMMA_BF16(a, b2, c2);
    c3 = WMMA_BF16(a, b3, c3);

    // C layout: element i, lane l -> M = i + 8*(l>>4), N = l&15
    int n   = lane & 15;
    int hiM = (lane >> 4) * 8;
#pragma unroll
    for (int j = 0; j < 4; ++j) {
        const v8f& c = (j == 0) ? c0 : (j == 1) ? c1 : (j == 2) ? c2 : c3;
        int   nj   = n0 + j * 16 + n;
        float bias = b_ih[nj];
#pragma unroll
        for (int i = 0; i < 8; ++i) {
            int M = row0 + hiM + i;
            gi[(long)M * G3 + nj] = c[i] + bias;
        }
    }
}

// ---------------------------------------------------------------------------
// Phase 2: one fused GRU step. grid = 32 blocks (32-wide HID slices),
// block = 128 threads = 4 waves. Wave tile: 32 rows x 16 cols x 3 gates
// = 6 wmma chains sharing 2 A fragments; 10 b128 loads / 6 wmma per k-iter,
// pipelined depth 2.
// ---------------------------------------------------------------------------
__global__ void __launch_bounds__(128)
gru_step_kernel(const __bf16* __restrict__ hbf_in,      // (64,1024) bf16
                const float* __restrict__ hprev_f32,    // (64,1024) f32
                const float* __restrict__ gi_t,         // (64,3072) f32
                const __bf16* __restrict__ Whb,         // (3072,1024) bf16
                const float* __restrict__ b_hh,         // (3072,)
                float* __restrict__ hout,               // (64,1024) f32
                unsigned short* __restrict__ hbf_out)   // (64,1024) bf16
{
    int wave = threadIdx.x >> 5;
    int lane = threadIdx.x & 31;
    int j0   = blockIdx.x * 32 + (wave & 1) * 16;  // HID col slice (16 wide)
    int m0   = (wave >> 1) * 32;                   // 32 batch rows

    const __bf16* A0 = hbf_in + (long)m0 * HID;
    const __bf16* A1 = hbf_in + (long)(m0 + 16) * HID;

    v8f c0r = {}, c0z = {}, c0n = {};   // rows m0..m0+15
    v8f c1r = {}, c1z = {}, c1n = {};   // rows m0+16..m0+31

    // pipeline prologue
    bf16x16 a0 = load_a_bf_rows(A0, HID, 0, lane);
    bf16x16 a1 = load_a_bf_rows(A1, HID, 0, lane);
    bf16x16 br = load_b_bf(Whb, HID, j0,           0, lane);
    bf16x16 bz = load_b_bf(Whb, HID, HID + j0,     0, lane);
    bf16x16 bn = load_b_bf(Whb, HID, 2 * HID + j0, 0, lane);

#pragma unroll 2
    for (int k0 = 0; k0 < HID - 32; k0 += 32) {
        int kn = k0 + 32;
        bf16x16 na0 = load_a_bf_rows(A0, HID, kn, lane);
        bf16x16 na1 = load_a_bf_rows(A1, HID, kn, lane);
        bf16x16 nbr = load_b_bf(Whb, HID, j0,           kn, lane);
        bf16x16 nbz = load_b_bf(Whb, HID, HID + j0,     kn, lane);
        bf16x16 nbn = load_b_bf(Whb, HID, 2 * HID + j0, kn, lane);
        c0r = WMMA_BF16(a0, br, c0r);
        c0z = WMMA_BF16(a0, bz, c0z);
        c0n = WMMA_BF16(a0, bn, c0n);
        c1r = WMMA_BF16(a1, br, c1r);
        c1z = WMMA_BF16(a1, bz, c1z);
        c1n = WMMA_BF16(a1, bn, c1n);
        a0 = na0; a1 = na1; br = nbr; bz = nbz; bn = nbn;
        SCHED_VMEM_READS(10);   // next iter's loads ahead of ...
        SCHED_WMMAS(6);         // ... this iter's 6 wmmas
    }
    c0r = WMMA_BF16(a0, br, c0r);
    c0z = WMMA_BF16(a0, bz, c0z);
    c0n = WMMA_BF16(a0, bn, c0n);
    c1r = WMMA_BF16(a1, br, c1r);
    c1z = WMMA_BF16(a1, bz, c1z);
    c1n = WMMA_BF16(a1, bn, c1n);

    int   n   = lane & 15;
    int   hiM = (lane >> 4) * 8;
    float bhr = b_hh[j0 + n];
    float bhz = b_hh[HID + j0 + n];
    float bhn = b_hh[2 * HID + j0 + n];

#pragma unroll
    for (int half = 0; half < 2; ++half) {
        const v8f& vr = half ? c1r : c0r;
        const v8f& vz = half ? c1z : c0z;
        const v8f& vn = half ? c1n : c0n;
        int mbase = m0 + half * 16 + hiM;
#pragma unroll
        for (int i = 0; i < 8; ++i) {
            int  M     = mbase + i;               // batch row
            long gbase = (long)M * G3 + j0 + n;
            float gir = gi_t[gbase];
            float giz = gi_t[gbase + HID];
            float gin = gi_t[gbase + 2 * HID];
            float hp  = hprev_f32[(long)M * HID + j0 + n];

            float r  = sigmoidf_(gir + vr[i] + bhr);
            float z  = sigmoidf_(giz + vz[i] + bhz);
            float nn = tanhf(gin + r * (vn[i] + bhn));
            float hn = (1.0f - z) * nn + z * hp;

            long oidx = (long)M * HID + j0 + n;
            hout[oidx]    = hn;
            hbf_out[oidx] = (unsigned short)(__builtin_bit_cast(unsigned, hn) >> 16);
        }
    }
}

// ---------------------------------------------------------------------------
extern "C" void kernel_launch(void* const* d_in, const int* in_sizes, int n_in,
                              void* d_out, int out_size, void* d_ws, size_t ws_size,
                              hipStream_t stream) {
    (void)in_sizes; (void)n_in; (void)out_size;
    const int*   input  = (const int*)  d_in[0];
    /* d_in[1] = input_lengths: unused by the reference output */
    const float* hidden = (const float*)d_in[2];   // (1,64,1024) zeros = h0
    const float* emb    = (const float*)d_in[3];
    const float* W_ih   = (const float*)d_in[4];
    const float* W_hh   = (const float*)d_in[5];
    const float* b_ih   = (const float*)d_in[6];
    const float* b_hh   = (const float*)d_in[7];
    float*       out    = (float*)d_out;           // (512,64,1024)

    // ---- workspace layout (all offsets naturally aligned) -----------------
    char* ws = (char*)d_ws;
    const size_t W_BYTES   = (size_t)G3 * HID * 2;          // 6 MB  (bf16)
    const size_t H_BYTES   = (size_t)BATCH * HID * 2;       // 128 KB (bf16)
    const size_t EMB_BYTES = (size_t)VOCAB * EMB_D * 2;     // 65.5 MB (bf16)
    const size_t STEP_GI   = (size_t)BATCH * G3 * 4;        // 768 KB (f32)

    __bf16* Wih_bf = (__bf16*)(ws);
    __bf16* Whh_bf = (__bf16*)(ws + W_BYTES);
    __bf16* hbf0   = (__bf16*)(ws + 2 * W_BYTES);
    __bf16* hbf1   = (__bf16*)(ws + 2 * W_BYTES + H_BYTES);
    size_t  fixed  = 2 * W_BYTES + 2 * H_BYTES;
    __bf16* emb_bf = (__bf16*)(ws + fixed);
    fixed += EMB_BYTES;                                     // ~78 MB fixed
    float*  gi     = (float*)(ws + fixed);

    // ---- phase 0: pack statics to bf16 ------------------------------------
    cvt_bf16_kernel<<<2048, 256, 0, stream>>>((const float2*)W_ih,
                                              (unsigned*)Wih_bf, (long)G3 * HID / 2);
    cvt_bf16_kernel<<<2048, 256, 0, stream>>>((const float2*)W_hh,
                                              (unsigned*)Whh_bf, (long)G3 * HID / 2);
    cvt_bf16_kernel<<<4096, 256, 0, stream>>>((const float2*)emb,
                                              (unsigned*)emb_bf, (long)VOCAB * EMB_D / 2);
    cvt_bf16_kernel<<<64, 256, 0, stream>>>((const float2*)hidden,
                                            (unsigned*)hbf0, (long)BATCH * HID / 2);

    // ---- chunk the time axis so the gi scratch fits remaining ws ----------
    long avail   = (long)ws_size - (long)fixed;
    int  t_chunk = (int)(avail / (long)STEP_GI);
    if (t_chunk < 1)   t_chunk = 1;
    if (t_chunk > SEQ) t_chunk = SEQ;

    __bf16* hbf[2] = { hbf0, hbf1 };

    for (int t0 = 0; t0 < SEQ; t0 += t_chunk) {
        int T    = SEQ - t0; if (T > t_chunk) T = t_chunk;
        int rows = T * BATCH;

        // phase 1 for this chunk: wave-tiles = (rows/16)*(3072/64), 8/block
        long tiles  = (long)(rows / 16) * (G3 / 64);
        int  blocks = (int)(tiles / 8);
        gi_kernel<<<blocks, 256, 0, stream>>>(input + (long)t0 * BATCH,
                                              emb_bf, Wih_bf, b_ih, gi);

        // phase 2: serial fused steps; bf16 h ping-pongs, f32 h via d_out.
        for (int t = t0; t < t0 + T; ++t) {
            const __bf16* hin   = hbf[t & 1];
            __bf16*       hnext = hbf[(t + 1) & 1];
            const float*  hpf   = (t == 0) ? hidden
                                           : out + (long)(t - 1) * BATCH * HID;
            gru_step_kernel<<<HID / 32 /* 32 */, 128, 0, stream>>>(
                hin, hpf,
                gi + (long)(t - t0) * BATCH * G3,
                Whh_bf, b_hh,
                out + (long)t * BATCH * HID,
                (unsigned short*)hnext);
        }
    }
}
